// HybridWindowTransformerLayer_83889301226220
// MI455X (gfx1250) — matrix operand order repaired and checked by reading, not compiled, not run on hardware
//
#include <hip/hip_runtime.h>

// ---------------------------------------------------------------------------
// HybridWindowTransformerLayer for MI455X (gfx1250, wave32, WMMA).
// All heavy math = K=512 GEMMs via v_wmma_f32_16x16x32_bf16.
// GEMM: waves share the A panel through LDS (32 rows x K=512 bf16, padded),
// staged via async global->LDS copies (ASYNCcnt). Waves split the N dim
// (2 row-groups x 4 col-groups, 16x128 output per wave, 8 accumulators).
// Per k-step: preload all 8 B fragments, then 8 back-to-back WMMAs.
// WIN=2 attention is a trivial per-pair softmax kernel.
// Rolls are folded into read-shift indices (T=4096 -> mask 0xFFF).
// ---------------------------------------------------------------------------

typedef __bf16 bf16;
typedef __attribute__((ext_vector_type(16))) __bf16 v16bf;
typedef __attribute__((ext_vector_type(8)))  __bf16 bf16x8;
typedef __attribute__((ext_vector_type(8)))  float  v8f;
typedef __attribute__((ext_vector_type(4)))  int    v4i;

#define NT_TOK 65536          // B*T = 16*4096 tokens
#define DMODEL 512
#define TMASK  4095           // T-1 (roll mask within a sequence)
#define TBITS  12             // log2(T)
#define KDIM   512
#define AROWS  32             // A-panel rows per workgroup
#define LDSB   520            // LDS row stride in bf16 (1040B == 4 dwords mod 64 banks)

#if __has_builtin(__builtin_amdgcn_global_load_async_to_lds_b128) && \
    __has_builtin(__builtin_amdgcn_s_wait_asynccnt)
#define USE_ASYNC_LDS 1
#endif

// ---------------------------------------------------------------------------
// Pack fp32 weight (K=512 x N, row major) into bf16 WMMA B-fragment layout.
// Block (ntile, kt) -> 32 lanes x 16 bf16, lane L: n = ntile*16 + (L&15),
// k = kt*32 + ((L>=16)?16:0) + e  (e = 0..15, contiguous).
// Packed blk index = ntile*16 + kt; lane data contiguous (32B per lane).
// ---------------------------------------------------------------------------
__global__ __launch_bounds__(256)
void k_pack_w(const float* __restrict__ W, int N, bf16* __restrict__ P) {
  int tid  = blockIdx.x * 256 + threadIdx.x;
  int lane = tid & 31;
  int blk  = tid >> 5;
  int nblocks = 16 * (N >> 4);            // (K/32) * (N/16)
  if (blk >= nblocks) return;
  int kt    = blk & 15;
  int ntile = blk >> 4;
  int n  = ntile * 16 + (lane & 15);
  int kb = kt * 32 + ((lane >> 4) ? 16 : 0);
  bf16* dst = P + ((size_t)blk * 512) + (size_t)lane * 16;
#pragma unroll
  for (int e = 0; e < 16; ++e)
    dst[e] = (bf16)W[(size_t)(kb + e) * N + n];
}

// ---------------------------------------------------------------------------
// GEMM: C[NT x N] = A[NT x 512](bf16) @ Wpacked + bias (+ optional residual
// read with per-sequence roll `rshift`).
// Workgroup: 8 waves, output tile 32 rows x 512 cols; grid (NT/32, N/512).
// A panel staged once in LDS (shared by the 4 col-group waves per row-group).
// ---------------------------------------------------------------------------
template <typename TR>
__global__ __launch_bounds__(256)
void k_gemm(const bf16* __restrict__ A, const bf16* __restrict__ Wp,
            const float* __restrict__ bias,
            const TR* __restrict__ resid, int rshift,
            bf16* __restrict__ C, int N) {
  __shared__ __attribute__((aligned(16))) bf16 As[AROWS * LDSB];  // ~32.5 KB

  const int tid  = threadIdx.x;
  const int lane = tid & 31;
  const int w    = tid >> 5;
  const int m0   = blockIdx.x * AROWS;

  // ---- stage A panel: 32 rows x 512 bf16 (1KB/row), 256 thr x 16B x 8 it ----
#if USE_ASYNC_LDS
  {
    typedef __attribute__((address_space(1))) v4i gv4i;
    typedef __attribute__((address_space(3))) v4i lv4i;
    unsigned long long gb = (unsigned long long)(const void*)(A + (size_t)m0 * KDIM);
    unsigned int lb = (unsigned int)(unsigned long long)(const void*)As;
#pragma unroll
    for (int it = 0; it < 8; ++it) {
      int chunk = it * 256 + tid;
      int prow = chunk >> 6;              // 0..31
      int kseg = chunk & 63;              // 16B segment within the 1KB row
      __builtin_amdgcn_global_load_async_to_lds_b128(
          (gv4i*)(gb + (unsigned long long)(prow * 1024 + kseg * 16)),
          (lv4i*)(lb + (unsigned int)(prow * (LDSB * 2) + kseg * 16)),
          0, 0);
    }
    __builtin_amdgcn_s_wait_asynccnt(0);
  }
#else
  {
#pragma unroll
    for (int it = 0; it < 8; ++it) {
      int chunk = it * 256 + tid;
      int prow = chunk >> 6;
      int kseg = chunk & 63;
      *(bf16x8*)(As + prow * LDSB + kseg * 8) =
          *(const bf16x8*)(A + (size_t)m0 * KDIM + (size_t)prow * KDIM + kseg * 8);
    }
  }
#endif
  __syncthreads();

  // ---- compute: wave (rg, cg) does rows rg*16..+16, col tiles nt0..nt0+7 ----
  const int rg  = w >> 2;                 // 0..1
  const int cg  = w & 3;                  // 0..3
  const int nt0 = blockIdx.y * 32 + cg * 8;
  const int khalf = (lane >> 4) * 16;     // lanes 16-31 own K 16-31 (contiguous)
  const bf16* Alds = As + (rg * 16 + (lane & 15)) * LDSB;
  const bf16* Bbase = Wp + (size_t)nt0 * 16 * 512 + (size_t)lane * 16;

  v8f zero = {0.f, 0.f, 0.f, 0.f, 0.f, 0.f, 0.f, 0.f};
  v8f acc[8];
#pragma unroll
  for (int j = 0; j < 8; ++j) acc[j] = zero;

  for (int kt = 0; kt < 16; ++kt) {
    // preload all 8 B fragments for this k-step (clauses into 16 b128 loads)
    v16bf bfr[8];
#pragma unroll
    for (int j = 0; j < 8; ++j)
      bfr[j] = *(const v16bf*)(Bbase + ((size_t)j * 16 + kt) * 512);
    // A fragment from LDS
    bf16x8 lo = *(const bf16x8*)(Alds + kt * 32 + khalf);
    bf16x8 hi = *(const bf16x8*)(Alds + kt * 32 + 8 + khalf);
    v16bf a;
#pragma unroll
    for (int e = 0; e < 8; ++e) { a[e] = lo[e]; a[e + 8] = hi[e]; }
    // dense WMMA burst
#pragma unroll
    for (int j = 0; j < 8; ++j)
      acc[j] = __builtin_amdgcn_wmma_f32_16x16x32_bf16(
          false, a, false, bfr[j], (short)0, acc[j], false, false);
  }

  // Epilogue: C/D layout -> VGPR r: row +r (+8 for lanes 16-31), col lane&15
  const int hi8 = (lane >> 4) * 8;
#pragma unroll
  for (int j = 0; j < 8; ++j) {
    int n = (nt0 + j) * 16 + (lane & 15);
    float bv = bias[n];
#pragma unroll
    for (int r = 0; r < 8; ++r) {
      int row = m0 + rg * 16 + hi8 + r;
      float v = acc[j][r] + bv;
      if (resid) {                        // residual only used when N==512
        int t  = row & TMASK;
        int b_ = row >> TBITS;
        int tr = (t + rshift) & TMASK;
        v += (float)resid[(((size_t)b_ << TBITS) | tr) * DMODEL + n];
      }
      C[(size_t)row * N + n] = (bf16)v;
    }
  }
}

// ---------------------------------------------------------------------------
// LayerNorm over D=512. One wave per token (16 contiguous elems per lane,
// shuffle reduction). Reads row (t + rshift) mod T inside its sequence.
// Optional bf16 add-buffer fused after LN (for final `x_self + xc`).
// ---------------------------------------------------------------------------
template <typename TIn, typename TOut>
__global__ __launch_bounds__(256)
void k_ln(const TIn* __restrict__ in, int rshift,
          const float* __restrict__ g, const float* __restrict__ bta,
          const bf16* __restrict__ addb, TOut* __restrict__ out) {
  int lane = threadIdx.x & 31;
  int w    = threadIdx.x >> 5;
  size_t row = (size_t)blockIdx.x * 8 + w;
  int t  = (int)(row & TMASK);
  int b_ = (int)(row >> TBITS);
  size_t rrow = ((size_t)b_ << TBITS) | (size_t)((t + rshift) & TMASK);

  const TIn* p = in + rrow * DMODEL + (size_t)lane * 16;
  float x[16];
  float s = 0.f, ss = 0.f;
#pragma unroll
  for (int i = 0; i < 16; ++i) {
    x[i] = (float)p[i];
    s += x[i];
    ss += x[i] * x[i];
  }
#pragma unroll
  for (int off = 16; off > 0; off >>= 1) {
    s  += __shfl_xor(s,  off, 32);
    ss += __shfl_xor(ss, off, 32);
  }
  float mean = s * (1.f / DMODEL);
  float var  = ss * (1.f / DMODEL) - mean * mean;
  float inv  = rsqrtf(var + 1e-5f);

  const float* gg = g   + lane * 16;
  const float* bb = bta + lane * 16;
  TOut* op = out + row * DMODEL + (size_t)lane * 16;
  const bf16* ap = addb ? addb + row * DMODEL + (size_t)lane * 16 : nullptr;
#pragma unroll
  for (int i = 0; i < 16; ++i) {
    float v = (x[i] - mean) * inv * gg[i] + bb[i];
    if (ap) v += (float)ap[i];
    op[i] = (TOut)v;
  }
}

// ---------------------------------------------------------------------------
// Window attention, WIN=2: per (pair, head, i) thread computes two 64-dim
// dots, 2-way softmax (SCALE = 64^-0.5 = 0.125, TEMP=1), blends v0/v1.
// q/k/v given as (ptr, leading-dim, col-offset) so self & cross share it.
// ---------------------------------------------------------------------------
__global__ __launch_bounds__(256)
void k_attn(const bf16* __restrict__ q, int qld, int qc0,
            const bf16* __restrict__ k, int kld, int kc0,
            const bf16* __restrict__ v, int vld, int vc0,
            bf16* __restrict__ out) {
  size_t idx = (size_t)blockIdx.x * 256 + threadIdx.x;   // NT*H threads
  size_t win = idx >> 4;
  int h = (int)((idx >> 1) & 7);
  int i = (int)(idx & 1);
  size_t t0 = win * 2;

  const bf16x8* qp = (const bf16x8*)(q + (t0 + i) * (size_t)qld + qc0 + h * 64);
  const bf16x8* k0 = (const bf16x8*)(k + t0 * (size_t)kld + kc0 + h * 64);
  const bf16x8* k1 = (const bf16x8*)((const bf16*)k0 + kld);
  const bf16x8* v0 = (const bf16x8*)(v + t0 * (size_t)vld + vc0 + h * 64);
  const bf16x8* v1 = (const bf16x8*)((const bf16*)v0 + vld);

  float s0 = 0.f, s1 = 0.f;
#pragma unroll
  for (int j = 0; j < 8; ++j) {
    bf16x8 tq = qp[j], tk0 = k0[j], tk1 = k1[j];
#pragma unroll
    for (int e = 0; e < 8; ++e) {
      float qd = (float)tq[e];
      s0 += qd * (float)tk0[e];
      s1 += qd * (float)tk1[e];
    }
  }
  s0 *= 0.125f;
  s1 *= 0.125f;
  float m  = fmaxf(s0, s1);
  float e0 = __expf(s0 - m), e1 = __expf(s1 - m);
  float inv = 1.f / (e0 + e1);
  float a0 = e0 * inv, a1 = e1 * inv;

  bf16* op = out + (t0 + i) * (size_t)DMODEL + h * 64;
#pragma unroll
  for (int j = 0; j < 8; ++j) {
    bf16x8 tv0 = v0[j], tv1 = v1[j];
#pragma unroll
    for (int e = 0; e < 8; ++e)
      op[j * 8 + e] = (bf16)(a0 * (float)tv0[e] + a1 * (float)tv1[e]);
  }
}

// ---------------------------------------------------------------------------
// Host-side orchestration.
// ---------------------------------------------------------------------------
extern "C" void kernel_launch(void* const* d_in, const int* in_sizes, int n_in,
                              void* d_out, int out_size, void* d_ws, size_t ws_size,
                              hipStream_t stream) {
  (void)in_sizes; (void)n_in; (void)out_size; (void)ws_size;

  const float* x         = (const float*)d_in[0];
  const float* y         = (const float*)d_in[1];
  const float* wa_qkv_w  = (const float*)d_in[2];
  const float* wa_qkv_b  = (const float*)d_in[3];
  const float* wa_proj_w = (const float*)d_in[4];
  const float* wa_proj_b = (const float*)d_in[5];
  const float* wa_n1_g   = (const float*)d_in[6];
  const float* wa_n1_b   = (const float*)d_in[7];
  const float* wa_n2_g   = (const float*)d_in[8];
  const float* wa_n2_b   = (const float*)d_in[9];
  const float* wv_qkv_w  = (const float*)d_in[10];
  const float* wv_qkv_b  = (const float*)d_in[11];
  const float* wv_proj_w = (const float*)d_in[12];
  const float* wv_proj_b = (const float*)d_in[13];
  const float* wv_n1_g   = (const float*)d_in[14];
  const float* wv_n1_b   = (const float*)d_in[15];
  const float* wv_n2_g   = (const float*)d_in[16];
  const float* wv_n2_b   = (const float*)d_in[17];
  const float* cm_q_w    = (const float*)d_in[18];
  const float* cm_q_b    = (const float*)d_in[19];
  const float* cm_kv_w   = (const float*)d_in[20];
  const float* cm_kv_b   = (const float*)d_in[21];
  const float* cm_proj_w = (const float*)d_in[22];
  const float* cm_proj_b = (const float*)d_in[23];
  const float* cm_n1_g   = (const float*)d_in[24];
  const float* cm_n1_b   = (const float*)d_in[25];
  const float* cm_n2_g   = (const float*)d_in[26];
  const float* cm_n2_b   = (const float*)d_in[27];

  float* out0 = (float*)d_out;
  float* out1 = out0 + (size_t)NT_TOK * DMODEL;

  // ---- workspace carve (256B aligned) ----
  size_t off = 0;
  auto carve = [&](size_t bytes) -> char* {
    off = (off + 255) & ~(size_t)255;
    char* p = (char*)d_ws + off;
    off += bytes;
    return p;
  };
  bf16* pWaqkv  = (bf16*)carve((size_t)512 * 1536 * 2);
  bf16* pWaproj = (bf16*)carve((size_t)512 * 512  * 2);
  bf16* pWvqkv  = (bf16*)carve((size_t)512 * 1536 * 2);
  bf16* pWvproj = (bf16*)carve((size_t)512 * 512  * 2);
  bf16* pWq     = (bf16*)carve((size_t)512 * 512  * 2);
  bf16* pWkv    = (bf16*)carve((size_t)512 * 1024 * 2);
  bf16* pWcp    = (bf16*)carve((size_t)512 * 512  * 2);
  bf16* bufH    = (bf16*)carve((size_t)NT_TOK * 512  * 2);
  bf16* bufQKV  = (bf16*)carve((size_t)NT_TOK * 1536 * 2);
  bf16* bufO    = (bf16*)carve((size_t)NT_TOK * 512  * 2);
  bf16* bufT1   = (bf16*)carve((size_t)NT_TOK * 512  * 2);
  bf16* xs      = (bf16*)carve((size_t)NT_TOK * 512  * 2);
  bf16* ys      = (bf16*)carve((size_t)NT_TOK * 512  * 2);
  bf16* bufQ    = bufQKV;
  bf16* bufKV   = bufQKV + (size_t)NT_TOK * 512;

  const dim3 blk256(256);
  const int LN_GRID = NT_TOK / 8;            // 8192
  const int AT_GRID = (NT_TOK * 8) / 256;    // 2048
  const int GM_GRID = NT_TOK / AROWS;        // 2048

  auto packW = [&](const float* W, int N, bf16* P) {
    int nth = 16 * (N >> 4) * 32;
    k_pack_w<<<dim3((nth + 255) / 256), blk256, 0, stream>>>(W, N, P);
  };
  packW(wa_qkv_w, 1536, pWaqkv);  packW(wa_proj_w, 512, pWaproj);
  packW(wv_qkv_w, 1536, pWvqkv);  packW(wv_proj_w, 512, pWvproj);
  packW(cm_q_w,   512,  pWq);     packW(cm_kv_w,   1024, pWkv);
  packW(cm_proj_w, 512, pWcp);

  auto gemmN = [&](const bf16* A, const bf16* Wp, const float* bias, int N, bf16* C) {
    k_gemm<bf16><<<dim3(GM_GRID, N / 512), blk256, 0, stream>>>(
        A, Wp, bias, (const bf16*)nullptr, 0, C, N);
  };
  auto gemmResB = [&](const bf16* A, const bf16* Wp, const float* bias,
                      const bf16* res, int rs, bf16* C) {
    k_gemm<bf16><<<dim3(GM_GRID, 1), blk256, 0, stream>>>(
        A, Wp, bias, res, rs, C, 512);
  };
  auto gemmResF = [&](const bf16* A, const bf16* Wp, const float* bias,
                      const float* res, bf16* C) {
    k_gemm<float><<<dim3(GM_GRID, 1), blk256, 0, stream>>>(
        A, Wp, bias, res, 0, C, 512);
  };

  // ================= self_tf(x) -> xs (audio params wa) =================
  // block 1 (no shift)
  k_ln<float, bf16><<<LN_GRID, blk256, 0, stream>>>(x, 0, wa_n1_g, wa_n1_b, nullptr, bufH);
  gemmN(bufH, pWaqkv, wa_qkv_b, 1536, bufQKV);
  k_attn<<<AT_GRID, blk256, 0, stream>>>(bufQKV, 1536, 0, bufQKV, 1536, 512,
                                         bufQKV, 1536, 1024, bufO);
  gemmResF(bufO, pWaproj, wa_proj_b, x, bufH);
  k_ln<bf16, bf16><<<LN_GRID, blk256, 0, stream>>>(bufH, 0, wa_n2_g, wa_n2_b, nullptr, bufT1);
  // block 2: input rolled by -1 (read shift +1); output rolled back (+1 -> read -1)
  k_ln<bf16, bf16><<<LN_GRID, blk256, 0, stream>>>(bufT1, 1, wa_n1_g, wa_n1_b, nullptr, bufH);
  gemmN(bufH, pWaqkv, wa_qkv_b, 1536, bufQKV);
  k_attn<<<AT_GRID, blk256, 0, stream>>>(bufQKV, 1536, 0, bufQKV, 1536, 512,
                                         bufQKV, 1536, 1024, bufO);
  gemmResB(bufO, pWaproj, wa_proj_b, bufT1, 1, bufH);
  k_ln<bf16, bf16><<<LN_GRID, blk256, 0, stream>>>(bufH, -1, wa_n2_g, wa_n2_b, nullptr, xs);

  // ================= self_tf(y) -> ys (video params wv) =================
  k_ln<float, bf16><<<LN_GRID, blk256, 0, stream>>>(y, 0, wv_n1_g, wv_n1_b, nullptr, bufH);
  gemmN(bufH, pWvqkv, wv_qkv_b, 1536, bufQKV);
  k_attn<<<AT_GRID, blk256, 0, stream>>>(bufQKV, 1536, 0, bufQKV, 1536, 512,
                                         bufQKV, 1536, 1024, bufO);
  gemmResF(bufO, pWvproj, wv_proj_b, y, bufH);
  k_ln<bf16, bf16><<<LN_GRID, blk256, 0, stream>>>(bufH, 0, wv_n2_g, wv_n2_b, nullptr, bufT1);
  k_ln<bf16, bf16><<<LN_GRID, blk256, 0, stream>>>(bufT1, 1, wv_n1_g, wv_n1_b, nullptr, bufH);
  gemmN(bufH, pWvqkv, wv_qkv_b, 1536, bufQKV);
  k_attn<<<AT_GRID, blk256, 0, stream>>>(bufQKV, 1536, 0, bufQKV, 1536, 512,
                                         bufQKV, 1536, 1024, bufO);
  gemmResB(bufO, pWvproj, wv_proj_b, bufT1, 1, bufH);
  k_ln<bf16, bf16><<<LN_GRID, blk256, 0, stream>>>(bufH, -1, wv_n2_g, wv_n2_b, nullptr, ys);

  // ================= cross transformers (shared cm params) =================
  // cx = q-side input (also residual), cy = kv-side input, s = read shift.
  auto cross_block = [&](const bf16* cx, const bf16* cy, int s,
                         int out_rs, const bf16* addb, void* outp, bool out_f32) {
    k_ln<bf16, bf16><<<LN_GRID, blk256, 0, stream>>>(cx, s, cm_n1_g, cm_n1_b, nullptr, bufH);
    k_gemm<bf16><<<dim3(GM_GRID, 1), blk256, 0, stream>>>(
        bufH, pWq, cm_q_b, (const bf16*)nullptr, 0, bufQ, 512);
    k_ln<bf16, bf16><<<LN_GRID, blk256, 0, stream>>>(cy, s, cm_n1_g, cm_n1_b, nullptr, bufH);
    k_gemm<bf16><<<dim3(GM_GRID, 2), blk256, 0, stream>>>(
        bufH, pWkv, cm_kv_b, (const bf16*)nullptr, 0, bufKV, 1024);
    k_attn<<<AT_GRID, blk256, 0, stream>>>(bufQ, 512, 0, bufKV, 1024, 0,
                                           bufKV, 1024, 512, bufO);
    gemmResB(bufO, pWcp, cm_proj_b, cx, s, bufH);
    if (out_f32)
      k_ln<bf16, float><<<LN_GRID, blk256, 0, stream>>>(bufH, out_rs, cm_n2_g, cm_n2_b,
                                                        addb, (float*)outp);
    else
      k_ln<bf16, bf16><<<LN_GRID, blk256, 0, stream>>>(bufH, out_rs, cm_n2_g, cm_n2_b,
                                                       nullptr, (bf16*)outp);
  };

  // xc: out0 = xs + roll(cross_block(roll(c1,-1), roll(ys,-1)), +1)
  cross_block(xs, ys, 0, 0, nullptr, bufT1, false);
  cross_block(bufT1, ys, 1, -1, xs, out0, true);
  // yc: out1 = ys + ...
  cross_block(ys, xs, 0, 0, nullptr, bufT1, false);
  cross_block(bufT1, xs, 1, -1, ys, out1, true);
}
